// AttentionBlock_51187420234503
// MI455X (gfx1250) — matrix operand order repaired and checked by reading, not compiled, not run on hardware
//
#include <hip/hip_runtime.h>
#include <math.h>

typedef __attribute__((ext_vector_type(16))) __bf16 v16bf;
typedef __attribute__((ext_vector_type(8)))  __bf16 v8bf;
typedef __attribute__((ext_vector_type(8)))  float  v8f;

// ---------------------------------------------------------------------------
// WMMA fragment loaders for v_wmma_f32_16x16x32_bf16 (wave32).
// A (16x32, MxK): lanes 0-15 hold M=lane, elems 0..7 -> K 0..7, 8..15 -> K 16..23
//                 lanes 16-31 hold M=lane-16, K offset +8 (8..15 / 24..31)
// B (32x16, KxN): lanes 0-15 hold N=lane,  elems 0..7 -> K 0..7, 8..15 -> K 16..23
//                 lanes 16-31 hold N=lane-16, K offset +8 (8..15 / 24..31)
// C/D (16x16):    elem r -> M = r + (lane<16?0:8), N = lane%16
// ---------------------------------------------------------------------------
__device__ __forceinline__ v16bf frag_a(const __bf16* rowPtr, int k0, int laneHi) {
  const __bf16* p = rowPtr + k0 + laneHi * 8;
  union { v16bf v; v8bf h[2]; } u;
  u.h[0] = *reinterpret_cast<const v8bf*>(p);
  u.h[1] = *reinterpret_cast<const v8bf*>(p + 16);
  return u.v;
}
__device__ __forceinline__ v16bf frag_b(const __bf16* rowPtr, int k0, int laneHi) {
  const __bf16* p = rowPtr + k0 + laneHi * 8;
  union { v16bf v; v8bf h[2]; } u;
  u.h[0] = *reinterpret_cast<const v8bf*>(p);
  u.h[1] = *reinterpret_cast<const v8bf*>(p + 16);
  return u.v;
}

// ---------------------------------------------------------------------------
// Generic batched GEMM: C[m,n] = sum_k A[m,k] * Bt[n,k]  (Bt is N-major, i.e.
// the B operand stored with K contiguous per row -> perfect B-fragment loads).
// MODE 0: out bf16 row-major, + bias[n]
// MODE 1: out bf16 TRANSPOSED [n][m], + bias[n]   (used to produce V^T)
// MODE 2: out f32 row-major, * scale              (scores / final output)
// Block = 128 threads = 4 waves (2x2), wave tile 32Mx64N, block tile 64Mx128N.
// ---------------------------------------------------------------------------
template <int MODE>
__global__ __launch_bounds__(128)
void gemm_bf16_wmma(const __bf16* __restrict__ A,  long lda, long aBatch,
                    const __bf16* __restrict__ Bt, long ldb, long bBatch,
                    const float* __restrict__ bias,
                    void* __restrict__ outPtr, long ldo, long oBatch,
                    int K, float scale) {
  const int tid    = threadIdx.x;
  const int lane   = tid & 31;
  const int wid    = tid >> 5;
  const int laneLo = lane & 15;
  const int laneHi = lane >> 4;
  const int waveM  = wid >> 1;
  const int waveN  = wid & 1;
  const int z      = blockIdx.z;
  const int m0     = blockIdx.x * 64  + waveM * 32;
  const int n0     = blockIdx.y * 128 + waveN * 64;

  const __bf16* Az = A  + (size_t)z * aBatch;
  const __bf16* Bz = Bt + (size_t)z * bBatch;

  const __bf16* aRow0 = Az + (size_t)(m0 + laneLo) * lda;
  const __bf16* aRow1 = aRow0 + (size_t)16 * lda;
  const __bf16* bRow[4];
#pragma unroll
  for (int j = 0; j < 4; ++j)
    bRow[j] = Bz + (size_t)(n0 + j * 16 + laneLo) * ldb;

  v8f acc[2][4];
#pragma unroll
  for (int i = 0; i < 2; ++i)
#pragma unroll
    for (int j = 0; j < 4; ++j)
      acc[i][j] = (v8f)(0.0f);

  for (int k0 = 0; k0 < K; k0 += 32) {
    v16bf a0 = frag_a(aRow0, k0, laneHi);
    v16bf a1 = frag_a(aRow1, k0, laneHi);
#pragma unroll
    for (int j = 0; j < 4; ++j) {
      v16bf b = frag_b(bRow[j], k0, laneHi);
      acc[0][j] = __builtin_amdgcn_wmma_f32_16x16x32_bf16(
          false, a0, false, b, (short)0, acc[0][j], false, false);
      acc[1][j] = __builtin_amdgcn_wmma_f32_16x16x32_bf16(
          false, a1, false, b, (short)0, acc[1][j], false, false);
    }
  }

  // ---- epilogue ----
#pragma unroll
  for (int i = 0; i < 2; ++i) {
#pragma unroll
    for (int j = 0; j < 4; ++j) {
      const int n  = n0 + j * 16 + laneLo;
      const float bs = (MODE == 2) ? 0.0f : bias[n];
      if (MODE == 0) {
        __bf16* out = (__bf16*)outPtr + (size_t)z * oBatch;
#pragma unroll
        for (int r = 0; r < 8; ++r) {
          const int m = m0 + i * 16 + laneHi * 8 + r;
          out[(size_t)m * ldo + n] = (__bf16)(acc[i][j][r] + bs);
        }
      } else if (MODE == 1) {
        __bf16* out = (__bf16*)outPtr + (size_t)z * oBatch;
        v8bf pk;
#pragma unroll
        for (int r = 0; r < 8; ++r) pk[r] = (__bf16)(acc[i][j][r] + bs);
        const int mBase = m0 + i * 16 + laneHi * 8;   // 8 consecutive M -> 16B store
        *reinterpret_cast<v8bf*>(out + (size_t)n * ldo + mBase) = pk;
      } else {
        float* out = (float*)outPtr + (size_t)z * oBatch;
#pragma unroll
        for (int r = 0; r < 8; ++r) {
          const int m = m0 + i * 16 + laneHi * 8 + r;
          out[(size_t)m * ldo + n] = acc[i][j][r] * scale;
        }
      }
    }
  }
}

// ---------------------------------------------------------------------------
// Conversions and row softmax
// ---------------------------------------------------------------------------
__global__ void cvt_f32_bf16(const float* __restrict__ in, __bf16* __restrict__ out, int n) {
  int i = blockIdx.x * blockDim.x + threadIdx.x;
  const int stride = gridDim.x * blockDim.x;
  for (; i < n; i += stride) out[i] = (__bf16)in[i];
}

// in: [rows x cols] f32 row-major  ->  out: [cols x rows] bf16 (transposed)
__global__ void transpose_f32_bf16(const float* __restrict__ in, __bf16* __restrict__ out,
                                   int rows, int cols) {
  int i = blockIdx.x * blockDim.x + threadIdx.x;
  const int total = rows * cols;
  const int stride = gridDim.x * blockDim.x;
  for (; i < total; i += stride) {
    const int r = i / cols, c = i - r * cols;
    out[(size_t)c * rows + r] = (__bf16)in[i];
  }
}

// one 256-thread block per row of length 2048 (8 elems/thread)
__global__ __launch_bounds__(256)
void softmax_rows(const float* __restrict__ Sc, __bf16* __restrict__ P, int n) {
  const int tid = threadIdx.x;
  const size_t row = blockIdx.x;
  const float* src = Sc + row * n;
  __bf16* dst = P + row * n;
  __shared__ float red[256];
  float v[8];
  float mx = -3.4e38f;
#pragma unroll
  for (int j = 0; j < 8; ++j) { v[j] = src[tid + j * 256]; mx = fmaxf(mx, v[j]); }
  red[tid] = mx; __syncthreads();
  for (int s = 128; s > 0; s >>= 1) { if (tid < s) red[tid] = fmaxf(red[tid], red[tid + s]); __syncthreads(); }
  mx = red[0]; __syncthreads();
  float sum = 0.f;
#pragma unroll
  for (int j = 0; j < 8; ++j) { v[j] = __expf(v[j] - mx); sum += v[j]; }
  red[tid] = sum; __syncthreads();
  for (int s = 128; s > 0; s >>= 1) { if (tid < s) red[tid] += red[tid + s]; __syncthreads(); }
  const float inv = 1.0f / red[0];
#pragma unroll
  for (int j = 0; j < 8; ++j) dst[tid + j * 256] = (__bf16)(v[j] * inv);
}

// ---------------------------------------------------------------------------
extern "C" void kernel_launch(void* const* d_in, const int* in_sizes, int n_in,
                              void* d_out, int out_size, void* d_ws, size_t ws_size,
                              hipStream_t stream) {
  (void)in_sizes; (void)n_in; (void)out_size; (void)ws_size;
  const float* x  = (const float*)d_in[0];
  const float* Wq = (const float*)d_in[1];
  const float* bq = (const float*)d_in[2];
  const float* Wk = (const float*)d_in[3];
  const float* bk = (const float*)d_in[4];
  const float* Wv = (const float*)d_in[5];
  const float* bv = (const float*)d_in[6];
  float* out = (float*)d_out;

  const int  B = 4, S = 2048, D = 1024;
  const long M = (long)B * S;  // 8192 total rows

  // workspace layout (all offsets 16B-aligned by construction)
  char* ws = (char*)d_ws;
  __bf16* Xb  = (__bf16*)ws; ws += (size_t)M * D * 2;   // x  as bf16      [8192x1024]
  __bf16* Wtq = (__bf16*)ws; ws += (size_t)D * D * 2;   // Wq^T bf16       [1024x1024]
  __bf16* Wtk = (__bf16*)ws; ws += (size_t)D * D * 2;
  __bf16* Wtv = (__bf16*)ws; ws += (size_t)D * D * 2;
  __bf16* Qb  = (__bf16*)ws; ws += (size_t)M * D * 2;   // Q bf16 row-major
  __bf16* Kb  = (__bf16*)ws; ws += (size_t)M * D * 2;   // K bf16 row-major
  __bf16* Vt  = (__bf16*)ws; ws += (size_t)D * M * 2;   // V^T bf16 [1024x8192]
  float*  Sc  = (float*)ws;  ws += (size_t)B * S * S * 4; // scores f32
  __bf16* P   = (__bf16*)ws; ws += (size_t)B * S * S * 2; // probs bf16

  // 1) precision conversion (+ weight transpose so B-fragments are contiguous)
  cvt_f32_bf16<<<2048, 256, 0, stream>>>(x, Xb, (int)(M * D));
  transpose_f32_bf16<<<1024, 256, 0, stream>>>(Wq, Wtq, D, D);
  transpose_f32_bf16<<<1024, 256, 0, stream>>>(Wk, Wtk, D, D);
  transpose_f32_bf16<<<1024, 256, 0, stream>>>(Wv, Wtv, D, D);

  const dim3 blk(128);

  // 2) QKV projections: [8192x1024] = Xb @ W + b
  const dim3 gq(M / 64, D / 128, 1);
  gemm_bf16_wmma<0><<<gq, blk, 0, stream>>>(Xb, D, 0, Wtq, D, 0, bq, Qb, D, 0, D, 1.0f);
  gemm_bf16_wmma<0><<<gq, blk, 0, stream>>>(Xb, D, 0, Wtk, D, 0, bk, Kb, D, 0, D, 1.0f);
  gemm_bf16_wmma<1><<<gq, blk, 0, stream>>>(Xb, D, 0, Wtv, D, 0, bv, Vt, M, 0, D, 1.0f);

  // 3) scores[b] = Q[b] @ K[b]^T  (K row-major already K-contiguous per key)
  const dim3 gs(S / 64, S / 128, B);
  gemm_bf16_wmma<2><<<gs, blk, 0, stream>>>(Qb, D, (long)S * D,
                                            Kb, D, (long)S * D, nullptr,
                                            Sc, S, (long)S * S, D, 1.0f);

  // 4) row softmax (unscaled scores, per reference) -> bf16 probs
  softmax_rows<<<(int)M, 256, 0, stream>>>(Sc, P, S);

  // 5) out[b] = (P[b] @ V[b]) / sqrt(1024); V^T rows give K-contiguous B frags,
  //    batch selects a 2048-wide column slice of Vt (bBatch = 2048 elements).
  const dim3 go(S / 64, D / 128, B);
  gemm_bf16_wmma<2><<<go, blk, 0, stream>>>(P, S, (long)S * S,
                                            Vt, M, (long)S, nullptr,
                                            out, D, (long)S * D, S, 0.03125f);
}
